// GNN_ActorCriticPolicy_13331578487072
// MI455X (gfx1250) — compile-verified, matrix operand order checked by reading, not run on hardware
//
#include <hip/hip_runtime.h>
#include <hip/hip_bf16.h>

typedef __attribute__((ext_vector_type(16))) _Float16 v16h;
typedef __attribute__((ext_vector_type(8)))  float    v8f;
typedef __attribute__((ext_vector_type(4)))  unsigned int u32x4;

#define NW      8          // waves per block (256 threads, wave32)
#define BLOCK   256
#define NFRAG   24         // swizzled f16 B-fragments kept in LDS
#define FRAG_DW 256        // 32 lanes * 8 dwords per fragment

#define WMMA16(a, b, c) \
    __builtin_amdgcn_wmma_f32_16x16x32_f16(false, (a), false, (b), (short)0, (c), false, false)

__device__ __forceinline__ float fast_tanh(float x) {
#if __has_builtin(__builtin_amdgcn_tanhf)
    return __builtin_amdgcn_tanhf(x);          // gfx1250 v_tanh_f32 (TRANS op)
#else
    float e = __expf(2.0f * x);
    return 1.0f - 2.0f * __builtin_amdgcn_rcpf(e + 1.0f);
#endif
}

__device__ __forceinline__ unsigned int pack2h(float lo, float hi) {
    union { _Float16 h[2]; unsigned int u; } cv;
    cv.h[0] = (_Float16)lo; cv.h[1] = (_Float16)hi;
    return cv.u;
}

// Load one 32x16 f16 B-fragment (pre-swizzled in LDS): 8 contiguous dwords/lane.
__device__ __forceinline__ v16h load_frag16(const unsigned int* base, int lane) {
    union { v16h v; u32x4 q[2]; } r;
    const u32x4* p = (const u32x4*)(base + lane * 8);
    r.q[0] = p[0];
    r.q[1] = p[1];
    return r.v;
}

// Rebuild a 16x32 f16 A-fragment from row-major LDS staging ([16][72] halves).
// lane M (<16): K = kf*32 + {0..7, 16..23}; lane M+16: +8  (ISA 7.12.2 layout).
__device__ __forceinline__ v16h load_a_stage(const _Float16* st, int lane, int kf) {
    int row = lane & 15;
    int colbase = kf * 32 + ((lane >= 16) ? 8 : 0);
    const _Float16* p = st + row * 72 + colbase;
    union { v16h v; u32x4 q[2]; } r;
    r.q[0] = *(const u32x4*)p;          // K offsets 0..7
    r.q[1] = *(const u32x4*)(p + 16);   // K offsets 16..23
    return r.v;
}

__global__ void __launch_bounds__(BLOCK, 4)
fused_gnn_policy(const float* __restrict__ obs, const float* __restrict__ t1,
                 const float* __restrict__ t2,
                 const float* __restrict__ loop_w1, const float* __restrict__ W1,
                 const float* __restrict__ m_bias1, const float* __restrict__ h_bias1,
                 const float* __restrict__ loop_w2, const float* __restrict__ W2,
                 const float* __restrict__ m_bias2, const float* __restrict__ h_bias2,
                 const float* __restrict__ fe_w,    const float* __restrict__ fe_b,
                 const float* __restrict__ common_w, const float* __restrict__ common_b,
                 const float* __restrict__ actor_w,  const float* __restrict__ actor_b,
                 const float* __restrict__ action_w, const float* __restrict__ action_b,
                 const float* __restrict__ log_std,
                 const float* __restrict__ critic_w, const float* __restrict__ critic_b,
                 const float* __restrict__ value_w,  const float* __restrict__ value_b,
                 float* __restrict__ out, int B, int nPair)
{
    __shared__ __align__(16) unsigned int wf[NFRAG * FRAG_DW];  // 24 KB swizzled weights
    __shared__ __align__(16) _Float16 stage[NW][16 * 72];       // per-wave transpose staging
    __shared__ __align__(16) float ostage[NW][48];              // per-wave output staging

    const int tid  = threadIdx.x;
    const int lane = tid & 31;
    const int w    = tid >> 5;

    // ---- one-time per block: build f16 WMMA B-fragments in LDS ----
    // frag 0..3  : common_w (8x64, K zero-padded to 32), n-tile = frag
    // frag 4..11 : actor_w  (64x64), idx = t*2+kf
    // frag 12..19: critic_w (64x64)
    // frag 20..21: action_w (64x2, N zero-padded to 16), kf = frag-20
    // frag 22..23: value_w  (64x1, N zero-padded to 16), kf = frag-22
    for (int g = tid; g < NFRAG * FRAG_DW; g += BLOCK) {
        int frag = g >> 8, d = g & 255;
        int fl = d >> 3, j = d & 7;
        int nl = fl & 15;
        int k  = ((fl >= 16) ? 8 : 0) + ((j >= 4) ? 16 : 0) + 2 * (j & 3);  // even, 0..30
        float lo = 0.f, hi = 0.f;
        if (frag < 4) {
            int n = frag * 16 + nl;
            if (k < 8)     lo = common_w[k * 64 + n];
            if (k + 1 < 8) hi = common_w[(k + 1) * 64 + n];
        } else if (frag < 12) {
            int idx = frag - 4, t = idx >> 1, kf = idx & 1;
            int kk = kf * 32 + k, n = t * 16 + nl;
            lo = actor_w[kk * 64 + n]; hi = actor_w[(kk + 1) * 64 + n];
        } else if (frag < 20) {
            int idx = frag - 12, t = idx >> 1, kf = idx & 1;
            int kk = kf * 32 + k, n = t * 16 + nl;
            lo = critic_w[kk * 64 + n]; hi = critic_w[(kk + 1) * 64 + n];
        } else if (frag < 22) {
            int kf = frag - 20, kk = kf * 32 + k;
            if (nl < 2) { lo = action_w[kk * 2 + nl]; hi = action_w[(kk + 1) * 2 + nl]; }
        } else {
            int kf = frag - 22, kk = kf * 32 + k;
            if (nl == 0) { lo = value_w[kk]; hi = value_w[kk + 1]; }
        }
        wf[g] = pack2h(lo, hi);
    }
    __syncthreads();

    // log_probs is constant: diff==0 -> sum_a(-log_std[a]) - A*0.5*log(2*pi), A=2
    const float lpc = -(log_std[0] + log_std[1]) - 1.8378770664093453f;

    const int gwid   = blockIdx.x * NW + w;
    const int nWaves = gridDim.x * NW;
    const int nl = lane & 15;
    const int mh = (lane >= 16) ? 8 : 0;

    #pragma unroll 1
    for (int pt = gwid; pt < nPair; pt += nWaves) {
        const int row = pt * 32 + lane;   // one row per lane (32 rows per iter)

        // ---- coalesced per-row inputs ----
        const float4 oA = *(const float4*)(obs + (size_t)row * 8);      // obs[0..3]
        const float4 oC = *(const float4*)(obs + (size_t)row * 8 + 4);  // obs[4..7]
        const float2 ta = *(const float2*)(t1 + (size_t)row * 2);
        const float2 tb = *(const float2*)(t2 + (size_t)row * 2);

        // ---- tiny GNN front-end (per-lane scalar; weights are uniform s-loads) ----
        float hx[4] = { oC.z, ta.x, tb.x, oA.x };
        float hy[4] = { oC.w, ta.y, tb.y, oA.y };
        const int SE[6] = { 0, 0, 0, 1, 1, 2 };

        float msg[6][4];
        #pragma unroll
        for (int e = 0; e < 6; ++e) {
            int s = SE[e];
            #pragma unroll
            for (int o = 0; o < 4; ++o)
                msg[e][o] = fmaf(hx[s], W1[e * 8 + o],
                            fmaf(hy[s], W1[e * 8 + 4 + o], m_bias1[e * 4 + o]));
        }
        float x1[4][4];
        #pragma unroll
        for (int n = 0; n < 4; ++n) {
            #pragma unroll
            for (int o = 0; o < 4; ++o) {
                float agg = 0.f;
                if (n == 1) agg = msg[0][o];
                else if (n == 2) agg = msg[1][o] + msg[3][o];
                else if (n == 3) agg = msg[2][o] + msg[4][o] + msg[5][o];
                float lp = fmaf(hx[n], loop_w1[n * 8 + o], hy[n] * loop_w1[n * 8 + 4 + o]);
                x1[n][o] = fast_tanh(agg + h_bias1[n * 4 + o] + lp);
            }
        }
        float m2[6];
        #pragma unroll
        for (int e = 0; e < 6; ++e) {
            int s = SE[e];
            float acc = m_bias2[e];
            #pragma unroll
            for (int i = 0; i < 4; ++i) acc = fmaf(x1[s][i], W2[e * 4 + i], acc);
            m2[e] = acc;
        }
        float go[4];
        #pragma unroll
        for (int n = 0; n < 4; ++n) {
            float agg = 0.f;
            if (n == 1) agg = m2[0];
            else if (n == 2) agg = m2[1] + m2[3];
            else if (n == 3) agg = m2[2] + m2[4] + m2[5];
            float acc = agg + h_bias2[n];
            #pragma unroll
            for (int i = 0; i < 4; ++i) acc = fmaf(x1[n][i], loop_w2[n * 4 + i], acc);
            go[n] = fast_tanh(acc);
        }
        float f[8];
        #pragma unroll
        for (int o = 0; o < 4; ++o) {
            float acc = fe_b[o];
            #pragma unroll
            for (int i = 0; i < 4; ++i) acc = fmaf(go[i], fe_w[i * 4 + o], acc);
            f[o] = fast_tanh(acc);
        }
        f[4] = oA.z; f[5] = oA.w; f[6] = oC.x; f[7] = oC.y;   // obs[2..5]

        _Float16* st = stage[w];

        // ---- two 16-row WMMA sub-tiles per iteration (rolled: one pipeline
        // live at a time keeps VGPR pressure bounded) ----
        #pragma unroll 1
        for (int s = 0; s < 2; ++s) {
            const int subBase = pt * 32 + s * 16;

            // gather rows s*16.. into lanes 0..15 as an A-fragment (K=8, zero-pad to 32)
            const int srcLane = nl + 16 * s;
            float gv[8];
            #pragma unroll
            for (int k = 0; k < 8; ++k) gv[k] = __shfl(f[k], srcLane, 32);
            v16h fa;
            #pragma unroll
            for (int k = 0; k < 16; ++k) fa[k] = (_Float16)0.f;
            if (lane < 16) {
                #pragma unroll
                for (int k = 0; k < 8; ++k) fa[k] = (_Float16)gv[k];
            }

            // shared = tanh(features @ common_w + common_b)  -> staging
            #pragma unroll 2
            for (int t = 0; t < 4; ++t) {
                v16h b = load_frag16(wf + t * FRAG_DW, lane);
                float bias = common_b[t * 16 + nl];
                v8f c;
                #pragma unroll
                for (int r = 0; r < 8; ++r) c[r] = bias;
                c = WMMA16(fa, b, c);
                #pragma unroll
                for (int r = 0; r < 8; ++r)
                    st[(mh + r) * 72 + t * 16 + nl] = (_Float16)fast_tanh(c[r]);
            }
            v16h sa0 = load_a_stage(st, lane, 0);
            v16h sa1 = load_a_stage(st, lane, 1);

            // latent_pi = tanh(shared @ actor_w + actor_b) -> staging
            #pragma unroll 2
            for (int t = 0; t < 4; ++t) {
                float bias = actor_b[t * 16 + nl];
                v8f c;
                #pragma unroll
                for (int r = 0; r < 8; ++r) c[r] = bias;
                c = WMMA16(sa0, load_frag16(wf + (4 + t * 2 + 0) * FRAG_DW, lane), c);
                c = WMMA16(sa1, load_frag16(wf + (4 + t * 2 + 1) * FRAG_DW, lane), c);
                #pragma unroll
                for (int r = 0; r < 8; ++r)
                    st[(mh + r) * 72 + t * 16 + nl] = (_Float16)fast_tanh(c[r]);
            }
            v16h pa0 = load_a_stage(st, lane, 0);
            v16h pa1 = load_a_stage(st, lane, 1);

            // mean_actions = latent_pi @ action_w + action_b  (N padded to 16)
            {
                v8f c;
                float bias = (nl < 2) ? action_b[nl] : 0.f;
                #pragma unroll
                for (int r = 0; r < 8; ++r) c[r] = bias;
                c = WMMA16(pa0, load_frag16(wf + 20 * FRAG_DW, lane), c);
                c = WMMA16(pa1, load_frag16(wf + 21 * FRAG_DW, lane), c);
                if (nl < 2) {
                    #pragma unroll
                    for (int r = 0; r < 8; ++r) ostage[w][(mh + r) * 2 + nl] = c[r];
                }
            }

            // latent_vf = tanh(shared @ critic_w + critic_b) -> staging
            #pragma unroll 2
            for (int t = 0; t < 4; ++t) {
                float bias = critic_b[t * 16 + nl];
                v8f c;
                #pragma unroll
                for (int r = 0; r < 8; ++r) c[r] = bias;
                c = WMMA16(sa0, load_frag16(wf + (12 + t * 2 + 0) * FRAG_DW, lane), c);
                c = WMMA16(sa1, load_frag16(wf + (12 + t * 2 + 1) * FRAG_DW, lane), c);
                #pragma unroll
                for (int r = 0; r < 8; ++r)
                    st[(mh + r) * 72 + t * 16 + nl] = (_Float16)fast_tanh(c[r]);
            }
            v16h va0 = load_a_stage(st, lane, 0);
            v16h va1 = load_a_stage(st, lane, 1);

            // values = latent_vf @ value_w + value_b  (N padded to 16)
            {
                v8f c;
                float bias = (nl == 0) ? value_b[0] : 0.f;
                #pragma unroll
                for (int r = 0; r < 8; ++r) c[r] = bias;
                c = WMMA16(va0, load_frag16(wf + 22 * FRAG_DW, lane), c);
                c = WMMA16(va1, load_frag16(wf + 23 * FRAG_DW, lane), c);
                if (nl == 0) {
                    #pragma unroll
                    for (int r = 0; r < 8; ++r) ostage[w][32 + mh + r] = c[r];
                }
            }

            // ---- coalesced output stores via LDS bounce ----
            out[(size_t)subBase * 2 + lane] = ostage[w][lane];             // actions [B,2]
            if (lane < 16) {
                out[(size_t)2 * B + subBase + lane] = ostage[w][32 + lane]; // values [B,1]
                out[(size_t)3 * B + subBase + lane] = lpc;                  // log_probs [B]
            }
        }
    }
}

extern "C" void kernel_launch(void* const* d_in, const int* in_sizes, int n_in,
                              void* d_out, int out_size, void* d_ws, size_t ws_size,
                              hipStream_t stream) {
    const float* obs      = (const float*)d_in[0];
    const float* t1       = (const float*)d_in[1];
    const float* t2       = (const float*)d_in[2];
    const float* loop_w1  = (const float*)d_in[3];
    const float* W1       = (const float*)d_in[4];
    const float* m_bias1  = (const float*)d_in[5];
    const float* h_bias1  = (const float*)d_in[6];
    const float* loop_w2  = (const float*)d_in[7];
    const float* W2       = (const float*)d_in[8];
    const float* m_bias2  = (const float*)d_in[9];
    const float* h_bias2  = (const float*)d_in[10];
    const float* fe_w     = (const float*)d_in[11];
    const float* fe_b     = (const float*)d_in[12];
    const float* common_w = (const float*)d_in[13];
    const float* common_b = (const float*)d_in[14];
    const float* actor_w  = (const float*)d_in[15];
    const float* actor_b  = (const float*)d_in[16];
    const float* action_w = (const float*)d_in[17];
    const float* action_b = (const float*)d_in[18];
    const float* log_std  = (const float*)d_in[19];
    const float* critic_w = (const float*)d_in[20];
    const float* critic_b = (const float*)d_in[21];
    const float* value_w  = (const float*)d_in[22];
    const float* value_b  = (const float*)d_in[23];

    const int B = in_sizes[0] / 8;        // obs is [B, 8]
    const int nPair = B / 32;             // 32 rows per wave-iteration (B = 2^20)
    int blocks = (nPair + NW * 4 - 1) / (NW * 4);   // ~4 pair-tiles per wave
    if (blocks > 1024) blocks = 1024;
    if (blocks < 1) blocks = 1;

    fused_gnn_policy<<<blocks, BLOCK, 0, stream>>>(
        obs, t1, t2, loop_w1, W1, m_bias1, h_bias1, loop_w2, W2, m_bias2, h_bias2,
        fe_w, fe_b, common_w, common_b, actor_w, actor_b, action_w, action_b,
        log_std, critic_w, critic_b, value_w, value_b,
        (float*)d_out, B, nPair);
}
